// MultiHeadGAT_31610959298898
// MI455X (gfx1250) — matrix operand order repaired and checked by reading, not compiled, not run on hardware
//
#include <hip/hip_runtime.h>
#include <hip/hip_bf16.h>

typedef __attribute__((ext_vector_type(16))) __bf16 v16bf;
typedef __attribute__((ext_vector_type(8)))  __bf16 v8bf;
typedef __attribute__((ext_vector_type(8)))  float  v8f;

#define GAT_N       3072
#define GAT_IN      512
#define GAT_OUT     64
#define GAT_HEADS   8
#define NEG_SLOPE   0.2f

// ---------------------------------------------------------------------------
// Kernel 0: transpose W [H][K][O] f32 -> WT [H][O][K] bf16 (K-major, so the
// WMMA B operand is a contiguous 16-element bf16 run per lane).
// ---------------------------------------------------------------------------
__global__ void gat_wt(const float* __restrict__ W, __bf16* __restrict__ WT) {
    int t = blockIdx.x * blockDim.x + threadIdx.x;     // [H*O*K)
    if (t >= GAT_HEADS * GAT_OUT * GAT_IN) return;
    int k    = t & (GAT_IN - 1);
    int o    = (t >> 9) & (GAT_OUT - 1);
    int head = t >> 15;
    WT[t] = (__bf16)W[((size_t)head * GAT_IN + k) * GAT_OUT + o];
}

// ---------------------------------------------------------------------------
// Kernel 1: per-head projection  h = x @ W[h]  via bf16 WMMA 16x16x32.
// One wave per (head, 16-row M tile). grid=(24,8), block=256 (8 waves).
// A layout (16-bit 16x32): lane r=L%16 -> M=r; half hf=L/16:
//   elem i -> K = hf*16 + i (contiguous, matching B)
// B layout (32x16): lane col=L%16; elem i -> K = hf*16 + i (contiguous).
// D layout: lane elem v -> (M = v + 8*hf, col = L%16).
// Output written K(row)-major: hT[h][o][n] bf16 -> one 16B store per n-tile.
// ---------------------------------------------------------------------------
__global__ void gat_project(const float* __restrict__ x,
                            const __bf16* __restrict__ WT,
                            __bf16* __restrict__ hT) {
    const int lane = threadIdx.x & 31;
    const int wave = threadIdx.x >> 5;
    const int r    = lane & 15;
    const int hf   = lane >> 4;
    const int head = blockIdx.y;
    const int tileM = (blockIdx.x * 8 + wave) * 16;
    const int row   = tileM + r;

    const float* __restrict__ xrow = x + (size_t)row * GAT_IN;

    v8f acc[4] = {};

    for (int k = 0; k < GAT_IN; k += 32) {
        // A: one contiguous 16-float run -> 4x float4 loads + pack to bf16
        const float4 x0 = *(const float4*)(xrow + k + hf * 16);
        const float4 x1 = *(const float4*)(xrow + k + hf * 16 + 4);
        const float4 x2 = *(const float4*)(xrow + k + hf * 16 + 8);
        const float4 x3 = *(const float4*)(xrow + k + hf * 16 + 12);
        v16bf a;
        a[0]=(__bf16)x0.x; a[1]=(__bf16)x0.y; a[2]=(__bf16)x0.z; a[3]=(__bf16)x0.w;
        a[4]=(__bf16)x1.x; a[5]=(__bf16)x1.y; a[6]=(__bf16)x1.z; a[7]=(__bf16)x1.w;
        a[8]=(__bf16)x2.x; a[9]=(__bf16)x2.y; a[10]=(__bf16)x2.z; a[11]=(__bf16)x2.w;
        a[12]=(__bf16)x3.x; a[13]=(__bf16)x3.y; a[14]=(__bf16)x3.z; a[15]=(__bf16)x3.w;

#pragma unroll
        for (int t = 0; t < 4; ++t) {
            // B: contiguous 16 bf16 (32B) from K-major WT -> two 16B loads
            const __bf16* bp = WT + ((size_t)head * GAT_OUT + t * 16 + r) * GAT_IN
                                  + k + hf * 16;
            v8bf lo = *(const v8bf*)bp;
            v8bf hi = *(const v8bf*)(bp + 8);
            v16bf b;
#pragma unroll
            for (int i = 0; i < 8; ++i) { b[i] = lo[i]; b[8 + i] = hi[i]; }
            acc[t] = __builtin_amdgcn_wmma_f32_16x16x32_bf16(
                false, a, false, b, (short)0, acc[t], false, false);
        }
    }

    // D rows v=0..7 are contiguous node indices in the transposed layout:
    // hT[(head*64 + col)*N + tileM + 8*hf + v]  -> one packed 16B bf16 store.
#pragma unroll
    for (int t = 0; t < 4; ++t) {
        v8bf st;
#pragma unroll
        for (int v = 0; v < 8; ++v) st[v] = (__bf16)acc[t][v];
        *(v8bf*)(hT + ((size_t)head * GAT_OUT + t * 16 + r) * GAT_N
                    + tileM + 8 * hf) = st;
    }
}

// ---------------------------------------------------------------------------
// Kernel 2: attention coefficients from the transposed h:
//   f_src[h][n] = sum_o hT[h][o][n] * a_src[h][o]   (coalesced in n)
// ---------------------------------------------------------------------------
__global__ void gat_coefs(const __bf16* __restrict__ hT,
                          const float* __restrict__ a_src,
                          const float* __restrict__ a_dst,
                          float* __restrict__ f_src,
                          float* __restrict__ f_dst) {
    int t = blockIdx.x * blockDim.x + threadIdx.x;
    if (t >= GAT_HEADS * GAT_N) return;
    int head = t / GAT_N;
    int n    = t - head * GAT_N;
    const __bf16* __restrict__ hb = hT + (size_t)head * GAT_OUT * GAT_N + n;
    const float* __restrict__ as  = a_src + head * GAT_OUT;
    const float* __restrict__ ad  = a_dst + head * GAT_OUT;
    float s = 0.f, d = 0.f;
#pragma unroll
    for (int o = 0; o < GAT_OUT; ++o) {
        float v = (float)hb[(size_t)o * GAT_N];
        s += v * as[o];
        d += v * ad[o];
    }
    f_src[t] = s;
    f_dst[t] = d;
}

// masked softmax probability for one (row, j) logit
__device__ __forceinline__ float gat_p(float aj, float fj, float fs,
                                       float m, float inv_l) {
    float e = fs + fj;
    e = e > 0.f ? e : NEG_SLOPE * e;
    return (aj > 0.f) ? __expf(e - m) * inv_l : 0.f;
}

// ---------------------------------------------------------------------------
// Kernel 3: fused masked-softmax attention + aggregation.
// grid = 192 (16-row i-tiles), block = 256: wave w = head w; adjacency rows
// shared across the 8 waves through WGP caches / L2. Exact two-pass f32
// softmax (max, then sum), 1/l folded into the bf16 A operand, then 96
// j-tiles of bf16 WMMA with B fed by contiguous loads from hT.
// ---------------------------------------------------------------------------
__global__ void gat_aggregate(const float* __restrict__ adj,
                              const __bf16* __restrict__ hT,
                              const float* __restrict__ f_src,
                              const float* __restrict__ f_dst,
                              float* __restrict__ out) {
    const int lane = threadIdx.x & 31;
    const int head = threadIdx.x >> 5;     // 8 waves = 8 heads
    const int r    = lane & 15;
    const int hf   = lane >> 4;
    const int itile = blockIdx.x;
    const int row   = itile * 16 + r;

    const float fs = f_src[head * GAT_N + row];
    const float* __restrict__ fd   = f_dst + head * GAT_N;
    const float* __restrict__ arow = adj + (size_t)row * GAT_N;

    // speculative prefetch of this block's adjacency rows, spread over lanes
    __builtin_prefetch(arow + hf * 1536 + r * 96, 0, 1);

    const int j0 = hf * 1536;              // lane-halves split the j range

    // ---- pass 1a: exact row max over masked LeakyReLU logits ----
    float m = -1e30f;
    for (int j = j0; j < j0 + 1536; j += 4) {
        float4 av = *(const float4*)(arow + j);
        float4 fv = *(const float4*)(fd + j);
        float e;
        e = fs + fv.x; e = e > 0.f ? e : NEG_SLOPE * e; if (av.x > 0.f && e > m) m = e;
        e = fs + fv.y; e = e > 0.f ? e : NEG_SLOPE * e; if (av.y > 0.f && e > m) m = e;
        e = fs + fv.z; e = e > 0.f ? e : NEG_SLOPE * e; if (av.z > 0.f && e > m) m = e;
        e = fs + fv.w; e = e > 0.f ? e : NEG_SLOPE * e; if (av.w > 0.f && e > m) m = e;
    }
    m = fmaxf(m, __shfl_xor(m, 16, 32));

    // ---- pass 1b: exact row sum of exp(e - m) over edges ----
    float s = 0.f;
    for (int j = j0; j < j0 + 1536; j += 4) {
        float4 av = *(const float4*)(arow + j);
        float4 fv = *(const float4*)(fd + j);
        float e;
        e = fs + fv.x; e = e > 0.f ? e : NEG_SLOPE * e; if (av.x > 0.f) s += __expf(e - m);
        e = fs + fv.y; e = e > 0.f ? e : NEG_SLOPE * e; if (av.y > 0.f) s += __expf(e - m);
        e = fs + fv.z; e = e > 0.f ? e : NEG_SLOPE * e; if (av.z > 0.f) s += __expf(e - m);
        e = fs + fv.w; e = e > 0.f ? e : NEG_SLOPE * e; if (av.w > 0.f) s += __expf(e - m);
    }
    s += __shfl_xor(s, 16, 32);
    const float inv_l = 1.0f / s;          // self-loop guarantees s > 0

    // ---- pass 2: out[16 x 64] = softmax(att) @ h  via bf16 WMMA over j ----
    const __bf16* __restrict__ hh = hT + (size_t)head * GAT_OUT * GAT_N;
    v8f acc[4] = {};

    for (int jb = 0; jb < GAT_N; jb += 32) {
        // A operand: one contiguous j-run of 16 per lane (K = hf*16 + i, matching B)
        const int b0 = jb + hf * 16;
        const int b1 = jb + hf * 16 + 8;
        float4 a0 = *(const float4*)(arow + b0);
        float4 a1 = *(const float4*)(arow + b0 + 4);
        float4 a2 = *(const float4*)(arow + b1);
        float4 a3 = *(const float4*)(arow + b1 + 4);
        float4 f0 = *(const float4*)(fd + b0);
        float4 f1 = *(const float4*)(fd + b0 + 4);
        float4 f2 = *(const float4*)(fd + b1);
        float4 f3 = *(const float4*)(fd + b1 + 4);
        v16bf a;
        a[0]  = (__bf16)gat_p(a0.x, f0.x, fs, m, inv_l);
        a[1]  = (__bf16)gat_p(a0.y, f0.y, fs, m, inv_l);
        a[2]  = (__bf16)gat_p(a0.z, f0.z, fs, m, inv_l);
        a[3]  = (__bf16)gat_p(a0.w, f0.w, fs, m, inv_l);
        a[4]  = (__bf16)gat_p(a1.x, f1.x, fs, m, inv_l);
        a[5]  = (__bf16)gat_p(a1.y, f1.y, fs, m, inv_l);
        a[6]  = (__bf16)gat_p(a1.z, f1.z, fs, m, inv_l);
        a[7]  = (__bf16)gat_p(a1.w, f1.w, fs, m, inv_l);
        a[8]  = (__bf16)gat_p(a2.x, f2.x, fs, m, inv_l);
        a[9]  = (__bf16)gat_p(a2.y, f2.y, fs, m, inv_l);
        a[10] = (__bf16)gat_p(a2.z, f2.z, fs, m, inv_l);
        a[11] = (__bf16)gat_p(a2.w, f2.w, fs, m, inv_l);
        a[12] = (__bf16)gat_p(a3.x, f3.x, fs, m, inv_l);
        a[13] = (__bf16)gat_p(a3.y, f3.y, fs, m, inv_l);
        a[14] = (__bf16)gat_p(a3.z, f3.z, fs, m, inv_l);
        a[15] = (__bf16)gat_p(a3.w, f3.w, fs, m, inv_l);

#pragma unroll
        for (int t = 0; t < 4; ++t) {
            // B operand: contiguous 16 bf16 from K(j)-major hT -> two 16B loads
            const __bf16* bp = hh + ((size_t)(t * 16 + r)) * GAT_N + jb + hf * 16;
            v8bf lo = *(const v8bf*)bp;
            v8bf hi = *(const v8bf*)(bp + 8);
            v16bf b;
#pragma unroll
            for (int i = 0; i < 8; ++i) { b[i] = lo[i]; b[8 + i] = hi[i]; }
            acc[t] = __builtin_amdgcn_wmma_f32_16x16x32_bf16(
                false, a, false, b, (short)0, acc[t], false, false);
        }
    }

    // out: [N, HEADS*OUT], row = itile*16 + v + 8*hf, col = head*64 + t*16 + r
#pragma unroll
    for (int t = 0; t < 4; ++t)
#pragma unroll
        for (int v = 0; v < 8; ++v)
            out[(size_t)(itile * 16 + v + 8 * hf) * (GAT_HEADS * GAT_OUT)
                + head * GAT_OUT + t * 16 + r] = acc[t][v];
}

// ---------------------------------------------------------------------------
extern "C" void kernel_launch(void* const* d_in, const int* in_sizes, int n_in,
                              void* d_out, int out_size, void* d_ws, size_t ws_size,
                              hipStream_t stream) {
    (void)in_sizes; (void)n_in; (void)out_size; (void)ws_size;

    const float* x     = (const float*)d_in[0];   // [3072, 512]
    const float* adj   = (const float*)d_in[1];   // [3072, 3072]
    const float* W     = (const float*)d_in[2];   // [8, 512, 64]
    const float* a_src = (const float*)d_in[3];   // [8, 64]
    const float* a_dst = (const float*)d_in[4];   // [8, 64]
    float* out = (float*)d_out;                   // [3072, 512]

    // workspace layout
    __bf16* hT = (__bf16*)d_ws;                                        // 8*64*3072 bf16 = 3 MiB
    __bf16* WT = hT + (size_t)GAT_HEADS * GAT_OUT * GAT_N;             // 8*64*512 bf16 = 0.5 MiB
    float*  f_src = (float*)(WT + (size_t)GAT_HEADS * GAT_OUT * GAT_IN);
    float*  f_dst = f_src + GAT_HEADS * GAT_N;

    // K0: W -> K-major bf16
    gat_wt<<<(GAT_HEADS * GAT_OUT * GAT_IN + 255) / 256, 256, 0, stream>>>(W, WT);

    // K1: projection (8 heads x 192 M-tiles; 8 waves/block), writes hT
    gat_project<<<dim3(24, GAT_HEADS), 256, 0, stream>>>(x, WT, hT);

    // K2: attention coefficients
    gat_coefs<<<(GAT_HEADS * GAT_N + 255) / 256, 256, 0, stream>>>(
        hT, a_src, a_dst, f_src, f_dst);

    // K3: fused masked softmax + aggregation (192 i-tiles; wave = head)
    gat_aggregate<<<GAT_N / 16, 256, 0, stream>>>(adj, hT, f_src, f_dst, out);
}
